// MultiHeadAttention_10239202034286
// MI455X (gfx1250) — matrix operand order repaired and checked by reading, not compiled, not run on hardware
//
#include <hip/hip_runtime.h>

// ---------------------------------------------------------------------------
// MHA forward for MI455X (gfx1250, wave32, WMMA + Tensor Data Mover).
// Pipeline:
//   f32->f16 convert (inputs + weights)
//   Q/K/V = x @ W^T + b  (f16 out)   gemm_nt_wmma<half>, TDM double-buffered
//   attn: flash-style causal softmax(QK^T/8)V per (b,h)  (f16 out), TDM K-tiles
//   out = cat @ Wo^T + bo (f32 out)  gemm_nt_wmma<float>
// ---------------------------------------------------------------------------

typedef _Float16 half_t;
typedef __attribute__((ext_vector_type(16))) _Float16 v16h;
typedef __attribute__((ext_vector_type(8)))  _Float16 v8h;
typedef __attribute__((ext_vector_type(4)))  _Float16 v4h;
typedef __attribute__((ext_vector_type(8)))  float    v8f;
typedef __attribute__((ext_vector_type(4)))  float    v4f;
typedef __attribute__((ext_vector_type(4)))  unsigned u32x4;
typedef __attribute__((ext_vector_type(8)))  int      i32x8;
typedef __attribute__((ext_vector_type(4)))  int      i32x4;

#define WMMA_F16(a, b, c) \
  __builtin_amdgcn_wmma_f32_16x16x32_f16(false, (a), false, (b), (short)0, (c), false, false)

static __device__ __forceinline__ v8f vzero8() {
  v8f r;
#pragma unroll
  for (int i = 0; i < 8; ++i) r[i] = 0.f;
  return r;
}

// A-fragment (16x32 f16, M x K): lane m=lane%16, half h=lane/16.
// lanes 0-15 hold K [0..7] and [16..23]; lanes 16-31 hold K [8..15] and [24..31].
static __device__ __forceinline__ v16h load_frag_a(const half_t* base, int stride) {
  const int lane = threadIdx.x & 31;
  const half_t* p = base + (lane & 15) * stride + (lane >> 4) * 8;
  v8h lo = *reinterpret_cast<const v8h*>(p);
  v8h hi = *reinterpret_cast<const v8h*>(p + 16);
  return __builtin_shufflevector(lo, hi, 0, 1, 2, 3, 4, 5, 6, 7, 8, 9, 10, 11, 12, 13, 14, 15);
}

// B-fragment (32x16 f16, K x N), source stored row-major as [N][K]:
// lane n=lane%16; lanes 0-15 hold K [0..7] and [16..23]; lanes 16-31 hold K [8..15] and [24..31].
static __device__ __forceinline__ v16h load_frag_b(const half_t* base, int stride) {
  const int lane = threadIdx.x & 31;
  const half_t* p = base + (lane & 15) * stride + (lane >> 4) * 8;
  v8h lo = *reinterpret_cast<const v8h*>(p);
  v8h hi = *reinterpret_cast<const v8h*>(p + 16);
  return __builtin_shufflevector(lo, hi, 0, 1, 2, 3, 4, 5, 6, 7, 8, 9, 10, 11, 12, 13, 14, 15);
}

// ---------------------------------------------------------------------------
// Tensor Data Mover: 2D tile (f16) global -> LDS, padded rows.
// Rows of tile_d0=64 halfs (32 DWORDs), pad 4 DWORDs -> LDS row stride 72 halfs
// (144B: 16B-aligned and bank-conflict-free for 16-row fragment gathers).
// g1 dw0: data_size=2B (1<<16) | pad_enable (1<<20) | pad_interval=32dw (4<<22)
//         | pad_amount=4dw (3<<25)
#define TDM_PAD72 0x07110000u

static __device__ __forceinline__ void tdm_load_2d(
    unsigned lds_off, const void* gaddr,
    unsigned tensor_d0, unsigned tensor_d1,
    unsigned tile_d0, unsigned tile_d1,
    unsigned stride_d0, unsigned cfg) {
  u32x4 g0;
  g0[0] = 1u;                                   // count=1 (valid user descriptor)
  g0[1] = lds_off;                              // LDS byte address
  unsigned long long ga = (unsigned long long)gaddr;
  g0[2] = (unsigned)ga;                         // global_addr[31:0]
  g0[3] = ((unsigned)(ga >> 32) & 0x01FFFFFFu) | 0x80000000u;  // addr[56:32] | type=2
  i32x8 g1;
  g1[0] = (int)cfg;                                            // mask=0 | data_size | pad
  g1[1] = (int)((tensor_d0 & 0xFFFFu) << 16);                  // bar_addr=0 | dim0 lo16
  g1[2] = (int)((tensor_d0 >> 16) | ((tensor_d1 & 0xFFFFu) << 16));
  g1[3] = (int)((tensor_d1 >> 16) | (tile_d0 << 16));          // dim1 hi16 | tile_dim0
  g1[4] = (int)(tile_d1 & 0xFFFFu);                            // tile_dim1 | tile_dim2=0
  g1[5] = (int)stride_d0;                                      // dim0_stride[31:0]
  g1[6] = 0;                                                   // stride hi | dim1_stride lo
  g1[7] = 0;
  i32x4 gz;
  gz[0] = 0; gz[1] = 0; gz[2] = 0; gz[3] = 0;                  // groups 2/3: 2D tensor
#if defined(__clang_major__) && (__clang_major__ >= 23)
  i32x8 gz8;
#pragma unroll
  for (int i = 0; i < 8; ++i) gz8[i] = 0;
  __builtin_amdgcn_tensor_load_to_lds(g0, g1, gz, gz, gz8, 0);
#else
  __builtin_amdgcn_tensor_load_to_lds(g0, g1, gz, gz, 0);
#endif
}

static __device__ __forceinline__ unsigned lds_off_of(const void* p) {
  return (unsigned)(unsigned long long)p;  // flat LDS aperture: addr[31:0] = LDS offset
}

// ---------------------------------------------------------------------------
__global__ void f32_to_f16_kernel(const float* __restrict__ in, half_t* __restrict__ out, int n4) {
  for (int i = blockIdx.x * blockDim.x + threadIdx.x; i < n4; i += gridDim.x * blockDim.x) {
    v4f x = reinterpret_cast<const v4f*>(in)[i];
    v4h y;
#pragma unroll
    for (int j = 0; j < 4; ++j) y[j] = (half_t)x[j];
    reinterpret_cast<v4h*>(out)[i] = y;
  }
}

// ---------------------------------------------------------------------------
// C[M,N] = A[M,K] @ Bw[N,K]^T + bias[N].  A, Bw row-major f16 (K contiguous).
// TDM double-buffered tile staging; BK=64 (2 WMMA K-chunks per barrier pair).
template <typename OutT>
__global__ __launch_bounds__(256) void gemm_nt_wmma(
    const half_t* __restrict__ A, const half_t* __restrict__ Bw,
    const float* __restrict__ bias, OutT* __restrict__ C,
    int M, int N, int K) {
  constexpr int BM = 128, BN = 128, BK = 64, LST = 72;
  __shared__ half_t As[2][BM * LST];
  __shared__ half_t Bs[2][BN * LST];

  const int tid  = threadIdx.x;
  const int lane = tid & 31;
  const int wid  = tid >> 5;
  const int ln   = lane & 15;
  const int hh   = lane >> 4;
  const int m0   = blockIdx.x * BM;
  const int n0   = blockIdx.y * BN;
  const int wm   = (wid & 3) * 32;   // wave tile: 32 (M) x 64 (N)
  const int wn   = (wid >> 2) * 64;

  v8f acc[2][4];
#pragma unroll
  for (int i = 0; i < 2; ++i)
#pragma unroll
    for (int j = 0; j < 4; ++j) acc[i][j] = vzero8();

  const half_t* Ab = A  + (size_t)m0 * K;
  const half_t* Bb = Bw + (size_t)n0 * K;
  const unsigned ldsA[2] = {lds_off_of(&As[0][0]), lds_off_of(&As[1][0])};
  const unsigned ldsB[2] = {lds_off_of(&Bs[0][0]), lds_off_of(&Bs[1][0])};

  if (wid == 0) {  // prologue: DMA tile 0 into buffer 0
    tdm_load_2d(ldsA[0], Ab, K, M, BK, BM, K, TDM_PAD72);
    tdm_load_2d(ldsB[0], Bb, K, N, BK, BN, K, TDM_PAD72);
  }

  int ib = 0;
  for (int kt = 0; kt < K; kt += BK, ib ^= 1) {
    if (wid == 0) {
      if (kt + BK < K) {  // DMA next tile into the other buffer, wait for current
        tdm_load_2d(ldsA[ib ^ 1], Ab + kt + BK, K, M, BK, BM, K, TDM_PAD72);
        tdm_load_2d(ldsB[ib ^ 1], Bb + kt + BK, K, N, BK, BN, K, TDM_PAD72);
        __builtin_amdgcn_s_wait_tensorcnt(2);
      } else {
        __builtin_amdgcn_s_wait_tensorcnt(0);
      }
    }
    __syncthreads();

    const half_t* as = &As[ib][0];
    const half_t* bs = &Bs[ib][0];
#pragma unroll
    for (int c = 0; c < 2; ++c) {
      // preload all fragments into distinct registers so DS loads overlap WMMA
      v16h af[2], bf[4];
#pragma unroll
      for (int im = 0; im < 2; ++im)
        af[im] = load_frag_a(as + (wm + im * 16) * LST + c * 32, LST);
#pragma unroll
      for (int jn = 0; jn < 4; ++jn)
        bf[jn] = load_frag_b(bs + (wn + jn * 16) * LST + c * 32, LST);
#pragma unroll
      for (int im = 0; im < 2; ++im)
#pragma unroll
        for (int jn = 0; jn < 4; ++jn)
          acc[im][jn] = WMMA_F16(af[im], bf[jn], acc[im][jn]);
    }
    __syncthreads();  // buffer ib fully consumed before it is re-targeted
  }

#pragma unroll
  for (int jn = 0; jn < 4; ++jn) {
    const int gn = n0 + wn + jn * 16 + ln;
    const float bb = bias[gn];
#pragma unroll
    for (int im = 0; im < 2; ++im) {
#pragma unroll
      for (int i = 0; i < 8; ++i) {
        const int gm = m0 + wm + im * 16 + i + 8 * hh;  // C layout: M = vgpr + 8*(lane/16)
        C[(size_t)gm * N + gn] = (OutT)(acc[im][jn][i] + bb);
      }
    }
  }
}

// ---------------------------------------------------------------------------
// Flash-style causal attention. Grid: (S/64, H, B). Block: 128 threads (4 waves).
// Wave owns 16 query rows; K tile via TDM, V^T (transposed) + P via manual LDS.
static constexpr int BB = 2, SS = 2048, EE = 1024, HH = 16, DH = 64;

__global__ __launch_bounds__(128) void attn_causal_wmma(
    const half_t* __restrict__ Q, const half_t* __restrict__ K,
    const half_t* __restrict__ V, half_t* __restrict__ O) {
  constexpr int TST = 72;
  __shared__ half_t Kt[64 * TST];
  __shared__ half_t VT[64 * TST];  // transposed: VT[d][k]
  __shared__ half_t Pt[64 * TST];  // 4 waves x 16 rows of P

  const int tid  = threadIdx.x;
  const int lane = tid & 31;
  const int wid  = tid >> 5;
  const int ln   = lane & 15;
  const int hh   = lane >> 4;
  const int qt = blockIdx.x, h = blockIdx.y, b = blockIdx.z;
  const int q0 = qt * 64 + wid * 16;

  const half_t* Qh = Q + (size_t)b * SS * EE + (size_t)h * DH;
  const half_t* Kh = K + (size_t)b * SS * EE + (size_t)h * DH;
  const half_t* Vh = V + (size_t)b * SS * EE + (size_t)h * DH;
  const unsigned ldsKt = lds_off_of(&Kt[0]);

  v16h qf[2];
#pragma unroll
  for (int c = 0; c < 2; ++c)
    qf[c] = load_frag_a(Qh + (size_t)q0 * EE + c * 32, EE);

  float mst[8], lst[8];
#pragma unroll
  for (int i = 0; i < 8; ++i) { mst[i] = -1e30f; lst[i] = 0.f; }
  v8f oacc[4];
#pragma unroll
  for (int dt = 0; dt < 4; ++dt) oacc[dt] = vzero8();

  const int srow = tid >> 1;          // V staging: 32 halfs per thread
  const int scol = (tid & 1) * 32;

  for (int j = 0; j <= qt; ++j) {
    const int k0 = j * 64;
    // --- K tile: DMA engine (row-major, padded to stride 72) ---
    if (wid == 0)
      tdm_load_2d(ldsKt, Kh + (size_t)k0 * EE, EE, SS, 64, 64, EE, TDM_PAD72);
    // --- V tile: manual transpose into VT[d][k] ---
    const half_t* vg = Vh + (size_t)(k0 + srow) * EE + scol;
#pragma unroll
    for (int p = 0; p < 4; ++p) {
      v8h vv = *reinterpret_cast<const v8h*>(vg + p * 8);
#pragma unroll
      for (int ii = 0; ii < 8; ++ii)
        VT[(scol + p * 8 + ii) * TST + srow] = vv[ii];
    }
    if (wid == 0) __builtin_amdgcn_s_wait_tensorcnt(0);
    __syncthreads();

    // --- scores = Q K^T (16x64 per wave); preload fragments per K-chunk ---
    v8f sa[4];
#pragma unroll
    for (int nt = 0; nt < 4; ++nt) sa[nt] = vzero8();
#pragma unroll
    for (int c = 0; c < 2; ++c) {
      v16h kf[4];
#pragma unroll
      for (int nt = 0; nt < 4; ++nt)
        kf[nt] = load_frag_b(&Kt[(nt * 16) * TST + c * 32], TST);
#pragma unroll
      for (int nt = 0; nt < 4; ++nt)
        sa[nt] = WMMA_F16(qf[c], kf[nt], sa[nt]);
    }

    // --- scale (1/sqrt(64)) + causal mask ---
#pragma unroll
    for (int nt = 0; nt < 4; ++nt) {
      const int kcol = k0 + nt * 16 + ln;
#pragma unroll
      for (int i = 0; i < 8; ++i) {
        const int qrow = q0 + i + 8 * hh;
        const float sv = sa[nt][i] * 0.125f;
        sa[nt][i] = (kcol <= qrow) ? sv : -1e30f;
      }
    }

    // --- online softmax (row stats across 16 lanes per half) ---
#pragma unroll
    for (int i = 0; i < 8; ++i) {
      float rm = -1e30f;
#pragma unroll
      for (int nt = 0; nt < 4; ++nt) rm = fmaxf(rm, sa[nt][i]);
#pragma unroll
      for (int off = 1; off < 16; off <<= 1) rm = fmaxf(rm, __shfl_xor(rm, off, 32));
      const float mnew  = fmaxf(mst[i], rm);
      const float alpha = __expf(mst[i] - mnew);
      float rs = 0.f;
#pragma unroll
      for (int nt = 0; nt < 4; ++nt) {
        const float p = __expf(sa[nt][i] - mnew);
        sa[nt][i] = p;
        rs += p;
      }
#pragma unroll
      for (int off = 1; off < 16; off <<= 1) rs += __shfl_xor(rs, off, 32);
      lst[i] = lst[i] * alpha + rs;
      mst[i] = mnew;
#pragma unroll
      for (int dt = 0; dt < 4; ++dt) oacc[dt][i] *= alpha;
    }

    // --- relayout P (C-layout f32 -> A-layout f16) via wave-private LDS ---
#pragma unroll
    for (int nt = 0; nt < 4; ++nt)
#pragma unroll
      for (int i = 0; i < 8; ++i)
        Pt[(wid * 16 + i + 8 * hh) * TST + nt * 16 + ln] = (half_t)sa[nt][i];
    asm volatile("s_wait_dscnt 0" ::: "memory");  // wave-local DS stores -> loads

    // --- O += P @ V; preload fragments per K-chunk ---
#pragma unroll
    for (int c = 0; c < 2; ++c) {
      v16h pf = load_frag_a(&Pt[(wid * 16) * TST + c * 32], TST);
      v16h vf[4];
#pragma unroll
      for (int dt = 0; dt < 4; ++dt)
        vf[dt] = load_frag_b(&VT[(dt * 16) * TST + c * 32], TST);
#pragma unroll
      for (int dt = 0; dt < 4; ++dt)
        oacc[dt] = WMMA_F16(pf, vf[dt], oacc[dt]);
    }
    __syncthreads();  // protect Kt/VT before next stage
  }

  // --- normalize + store concat[b, s, h*64 + d] ---
  half_t* Oh = O + (size_t)b * SS * EE + (size_t)h * DH;
#pragma unroll
  for (int dt = 0; dt < 4; ++dt)
#pragma unroll
    for (int i = 0; i < 8; ++i) {
      const int row = q0 + i + 8 * hh;
      Oh[(size_t)row * EE + dt * 16 + ln] = (half_t)(oacc[dt][i] / lst[i]);
    }
}

// ---------------------------------------------------------------------------
extern "C" void kernel_launch(void* const* d_in, const int* in_sizes, int n_in,
                              void* d_out, int out_size, void* d_ws, size_t ws_size,
                              hipStream_t stream) {
  (void)in_sizes; (void)n_in; (void)out_size; (void)ws_size;

  const float* q  = (const float*)d_in[0];
  const float* k  = (const float*)d_in[1];
  const float* v  = (const float*)d_in[2];
  // d_in[3] = mask (causal), computed analytically in-kernel
  const float* Wq = (const float*)d_in[4];
  const float* bq = (const float*)d_in[5];
  const float* Wk = (const float*)d_in[6];
  const float* bk = (const float*)d_in[7];
  const float* Wv = (const float*)d_in[8];
  const float* bv = (const float*)d_in[9];
  const float* Wo = (const float*)d_in[10];
  const float* bo = (const float*)d_in[11];
  float* out = (float*)d_out;

  const size_t tokE = (size_t)BB * SS * EE;  // 4,194,304
  const size_t wE   = (size_t)EE * EE;       // 1,048,576

  char* ws = (char*)d_ws;                    // total ~67 MB
  half_t* q16   = (half_t*)ws; ws += tokE * sizeof(half_t);
  half_t* k16   = (half_t*)ws; ws += tokE * sizeof(half_t);
  half_t* v16   = (half_t*)ws; ws += tokE * sizeof(half_t);
  half_t* Wq16  = (half_t*)ws; ws += wE * sizeof(half_t);
  half_t* Wk16  = (half_t*)ws; ws += wE * sizeof(half_t);
  half_t* Wv16  = (half_t*)ws; ws += wE * sizeof(half_t);
  half_t* Wo16  = (half_t*)ws; ws += wE * sizeof(half_t);
  half_t* Q16   = (half_t*)ws; ws += tokE * sizeof(half_t);
  half_t* K16   = (half_t*)ws; ws += tokE * sizeof(half_t);
  half_t* V16   = (half_t*)ws; ws += tokE * sizeof(half_t);
  half_t* cat16 = (half_t*)ws; ws += tokE * sizeof(half_t);

  f32_to_f16_kernel<<<1024, 256, 0, stream>>>(q,  q16,  (int)(tokE / 4));
  f32_to_f16_kernel<<<1024, 256, 0, stream>>>(k,  k16,  (int)(tokE / 4));
  f32_to_f16_kernel<<<1024, 256, 0, stream>>>(v,  v16,  (int)(tokE / 4));
  f32_to_f16_kernel<<<512,  256, 0, stream>>>(Wq, Wq16, (int)(wE / 4));
  f32_to_f16_kernel<<<512,  256, 0, stream>>>(Wk, Wk16, (int)(wE / 4));
  f32_to_f16_kernel<<<512,  256, 0, stream>>>(Wv, Wv16, (int)(wE / 4));
  f32_to_f16_kernel<<<512,  256, 0, stream>>>(Wo, Wo16, (int)(wE / 4));

  const int M = BB * SS;  // 4096
  dim3 gg(M / 128, EE / 128);
  gemm_nt_wmma<half_t><<<gg, 256, 0, stream>>>(q16, Wq16, bq, Q16, M, EE, EE);
  gemm_nt_wmma<half_t><<<gg, 256, 0, stream>>>(k16, Wk16, bk, K16, M, EE, EE);
  gemm_nt_wmma<half_t><<<gg, 256, 0, stream>>>(v16, Wv16, bv, V16, M, EE, EE);

  dim3 ga(SS / 64, HH, BB);
  attn_causal_wmma<<<ga, 128, 0, stream>>>(Q16, K16, V16, cat16);

  gemm_nt_wmma<float><<<gg, 256, 0, stream>>>(cat16, Wo16, bo, out, M, EE, EE);
}
